// Core_Adaptive_32933809226059
// MI455X (gfx1250) — matrix operand order repaired and checked by reading, not compiled, not run on hardware
//
#include <hip/hip_runtime.h>
#include <math.h>

// ---------------------------------------------------------------------------
// Shapes (compile-time constants from the reference)
// ---------------------------------------------------------------------------
#define BB 16
#define TT 12
#define NN 2048
#define FF 64
#define HEAD 4
#define HDIM 16
#define CORES 64
#define DFFN_IN 128
#define DFFN_HID 512
#define DOUT 64

typedef __attribute__((ext_vector_type(16))) __bf16 v16bf;
typedef __attribute__((ext_vector_type(8)))  float  v8f;

__device__ __forceinline__ unsigned short f2bf(float f) {
    union { float f; unsigned int u; } v; v.f = f;
    unsigned int u = v.u;
    unsigned int r = u + 0x7FFFu + ((u >> 16) & 1u);   // round-to-nearest-even
    return (unsigned short)(r >> 16);
}
__device__ __forceinline__ float bf2f(unsigned short h) {
    union { unsigned int u; float f; } v; v.u = ((unsigned int)h) << 16;
    return v.f;
}

// Branch-free exact-GELU: Abramowitz-Stegun 7.1.26 erf (max err 1.5e-7),
// v_rcp_f32 + v_exp_f32, no divergence (unlike libm erff's |x|<1 split).
__device__ __forceinline__ float gelu_f(float x) {
    float z  = x * 0.70710678118654752f;
    float az = fabsf(z);
    float t  = __builtin_amdgcn_rcpf(1.0f + 0.3275911f * az);
    float p  = ((((1.061405429f * t - 1.453152027f) * t + 1.421413741f) * t
                 - 0.284496736f) * t + 0.254829592f) * t;
    float e  = __expf(-az * az);
    float er = copysignf(1.0f - p * e, z);
    return 0.5f * x * (1.0f + er);
}

// ---------------------------------------------------------------------------
// Kernel 0: weight prep.  Convert ffn weights to bf16 in WMMA A-fragment
// order:  frag[((mt*KT + kt)*32 + lane)*16 + j]  holds  W[m][k] with
//   m = mt*16 + (lane&15)
//   k = kt*32 + j + ((lane>>4)<<4)      (contiguous k order, matches B fragments)
// Also zeroes the 128-float BN accumulator block.
// ---------------------------------------------------------------------------
__global__ __launch_bounds__(256) void prep_kernel(const float* __restrict__ w1,
                                                   const float* __restrict__ w2,
                                                   unsigned short* __restrict__ w1s,
                                                   unsigned short* __restrict__ w2s,
                                                   float* __restrict__ gsums) {
    int idx = blockIdx.x * 256 + threadIdx.x;
    if (idx < 128) gsums[idx] = 0.0f;
    if (idx < DFFN_HID * DFFN_IN) {           // 65536
        int j = idx & 15, l = (idx >> 4) & 31, frag = idx >> 9;
        int kt = frag & 3, mt = frag >> 2;    // K=128 -> 4 ksteps, M=512 -> 32 tiles
        int m = (mt << 4) + (l & 15);
        int k = (kt << 5) + j + ((l >> 4) << 4);
        w1s[idx] = f2bf(w1[m * DFFN_IN + k]);
    }
    int i2 = idx - DFFN_HID * DFFN_IN;
    if (i2 >= 0 && i2 < DOUT * DFFN_HID) {    // 32768
        int j = i2 & 15, l = (i2 >> 4) & 31, frag = i2 >> 9;
        int kt = frag & 15, mt = frag >> 4;   // K=512 -> 16 ksteps, M=64 -> 4 tiles
        int m = (mt << 4) + (l & 15);
        int k = (kt << 5) + j + ((l >> 4) << 4);
        w2s[i2] = f2bf(w2[m * DFFN_HID + k]);
    }
}

// ---------------------------------------------------------------------------
// Kernel 1: attention-like core routing.  One workgroup per (b,h,t).
// Streams N in 256-wide chunks; online softmax over nodes (A1) with rescaled
// U1 accumulation (all reductions 4-way parallel over 256 threads); 2nd pass
// recomputes affinities for the per-node softmax over cores (A2) and emits V2
// with the torch transpose/reshape scramble applied, stored bf16.
// sAff padded to 257 so bank index is (c + n) % 64 -> conflict-free patterns.
// ---------------------------------------------------------------------------
#define AT_NCH 256

__global__ __launch_bounds__(256) void attn_kernel(const float* __restrict__ in,
                                                   const float* __restrict__ cores,
                                                   const float* __restrict__ vw,
                                                   const float* __restrict__ vb,
                                                   unsigned short* __restrict__ vout) {
    __shared__ float sW[16][64];
    __shared__ float sBi[16];
    __shared__ float sC[64][16];
    __shared__ float sQ[16][AT_NCH + 1];
    __shared__ float sXh[16][AT_NCH + 1];
    __shared__ float sAff[64][AT_NCH + 1];
    __shared__ float sU1[16][64];
    __shared__ float sV1[16][64];
    __shared__ float sRed[64][4];
    __shared__ float sM[64], sL[64], sScale[64];

    const int g = blockIdx.x;
    const int b = g / (HEAD * TT);
    const int h = (g / TT) & (HEAD - 1);
    const int t = g % TT;
    const int tid = threadIdx.x;
    const int rc = tid & 63;          // core row for reductions
    const int rq = tid >> 6;          // quarter (whole wave shares rq)

    for (int i = tid; i < 16 * 64; i += 256) sW[i >> 6][i & 63] = vw[(h * 16 + (i >> 6)) * 64 + (i & 63)];
    if (tid < 16) sBi[tid] = vb[h * 16 + tid];
    for (int i = tid; i < 64 * 16; i += 256) sC[i >> 4][i & 15] = cores[(h * 64 + (i >> 4)) * 16 + (i & 15)];
    for (int i = tid; i < 1024; i += 256) sU1[i >> 6][i & 63] = 0.0f;
    if (tid < 64) { sM[tid] = -1e30f; sL[tid] = 0.0f; }
    __syncthreads();

    const long base_btn = ((long)(b * TT + t)) * NN;

    // ----------------- pass 1: A1 online softmax + U1 -----------------------
    for (int c0 = 0; c0 < NN; c0 += AT_NCH) {
        {   // q[d][n] and x_head[fd][n]
            const int n = tid;
            const float* xr = in + (base_btn + c0 + n) * FF;
            if (c0 + AT_NCH < NN) {   // warm next chunk (global_prefetch_b8)
                const char* pf = (const char*)(xr + AT_NCH * FF);
                __builtin_prefetch(pf, 0, 1);
                __builtin_prefetch(pf + 128, 0, 1);
            }
            float acc[16];
            #pragma unroll
            for (int d = 0; d < 16; ++d) acc[d] = sBi[d];
            for (int f = 0; f < 64; ++f) {
                float xv = xr[f];
                if ((f >> 4) == h) sXh[f & 15][n] = xv;
                #pragma unroll
                for (int d = 0; d < 16; ++d) acc[d] += sW[d][f] * xv;
            }
            #pragma unroll
            for (int d = 0; d < 16; ++d) sQ[d][n] = acc[d];
        }
        __syncthreads();
        {   // aff[c][n], every thread owns one n
            float qv[16];
            #pragma unroll
            for (int d = 0; d < 16; ++d) qv[d] = sQ[d][tid];
            for (int c = 0; c < 64; ++c) {
                float a = 0.0f;
                #pragma unroll
                for (int d = 0; d < 16; ++d) a += sC[c][d] * qv[d];
                sAff[c][tid] = a * 0.25f;       // 1/sqrt(16)
            }
        }
        __syncthreads();
        {   // parallel row-max partials (thread (rc,rq) scans 64 columns)
            float mc = -1e30f;
            for (int n = rq * 64; n < rq * 64 + 64; ++n) mc = fmaxf(mc, sAff[rc][n]);
            sRed[rc][rq] = mc;
        }
        __syncthreads();
        if (tid < 64) {                         // combine, update running max
            float mc = fmaxf(fmaxf(sRed[tid][0], sRed[tid][1]),
                             fmaxf(sRed[tid][2], sRed[tid][3]));
            float mn = fmaxf(sM[tid], mc);
            sScale[tid] = __expf(sM[tid] - mn);
            sM[tid] = mn;
        }
        __syncthreads();
        {   // exponentiate in place with new max, partial sums
            float mn = sM[rc];
            float ls = 0.0f;
            for (int n = rq * 64; n < rq * 64 + 64; ++n) {
                float e = __expf(sAff[rc][n] - mn);
                sAff[rc][n] = e;
                ls += e;
            }
            sRed[rc][rq] = ls;
        }
        __syncthreads();
        if (tid < 64)
            sL[tid] = sL[tid] * sScale[tid]
                    + (sRed[tid][0] + sRed[tid][1] + sRed[tid][2] + sRed[tid][3]);
        __syncthreads();
        for (int i = tid; i < 1024; i += 256) { // U1[f][c] rescale + accumulate
            int f = i >> 6, c = i & 63;
            float s = 0.0f;
            for (int n = 0; n < AT_NCH; ++n) s += sXh[f][n] * sAff[c][n];
            sU1[f][c] = sU1[f][c] * sScale[c] + s;
        }
        __syncthreads();
    }
    for (int i = tid; i < 1024; i += 256) {     // V1 = U1 / l
        int f = i >> 6, c = i & 63;
        sV1[f][c] = sU1[f][c] / sL[c];
    }
    __syncthreads();

    // ----------------- pass 2: A2 softmax over cores + V2 -------------------
    const int bprime = h * 4 + (b >> 2);
    for (int c0 = 0; c0 < NN; c0 += AT_NCH) {
        {   // recompute q
            const int n = tid;
            const float* xr = in + (base_btn + c0 + n) * FF;
            if (c0 + AT_NCH < NN) {
                const char* pf = (const char*)(xr + AT_NCH * FF);
                __builtin_prefetch(pf, 0, 1);
                __builtin_prefetch(pf + 128, 0, 1);
            }
            float acc[16];
            #pragma unroll
            for (int d = 0; d < 16; ++d) acc[d] = sBi[d];
            for (int f = 0; f < 64; ++f) {
                float xv = xr[f];
                #pragma unroll
                for (int d = 0; d < 16; ++d) acc[d] += sW[d][f] * xv;
            }
            #pragma unroll
            for (int d = 0; d < 16; ++d) sQ[d][n] = acc[d];
        }
        __syncthreads();
        {
            float qv[16];
            #pragma unroll
            for (int d = 0; d < 16; ++d) qv[d] = sQ[d][tid];
            for (int c = 0; c < 64; ++c) {
                float a = 0.0f;
                #pragma unroll
                for (int d = 0; d < 16; ++d) a += sC[c][d] * qv[d];
                sAff[c][tid] = a * 0.25f;
            }
        }
        __syncthreads();
        {   // per-column softmax over c fused with V2 = V1 * A2
            const int n = tid;
            float mx = -1e30f;
            for (int c = 0; c < 64; ++c) mx = fmaxf(mx, sAff[c][n]);
            float acc[16];
            #pragma unroll
            for (int f = 0; f < 16; ++f) acc[f] = 0.0f;
            float ssum = 0.0f;
            for (int c = 0; c < 64; ++c) {
                float e = __expf(sAff[c][n] - mx);
                ssum += e;
                #pragma unroll
                for (int f = 0; f < 16; ++f) acc[f] += sV1[f][c] * e;
            }
            float invs = 1.0f / ssum;
            #pragma unroll
            for (int fd = 0; fd < 16; ++fd) {   // scrambled store: (h,b,fd)->(b',f')
                int fp = ((b & 3) << 4) + fd;
                long idx = (((long)(bprime * 64 + fp)) * TT + t) * NN + c0 + n;
                vout[idx] = f2bf(acc[fd] * invs);
            }
        }
        __syncthreads();
    }
}

// ---------------------------------------------------------------------------
// Kernel 2: fused FFN (concat -> GEMM1(bf16 WMMA) -> exact GELU -> GEMM2
// (bf16 WMMA) -> +bias +residual).  Multi-accumulator form: each A-fragment
// (weights) is loaded once per k-step and issued against 4 LDS B-fragments
// back-to-back.  Writes pre-BN fp32 in (b,t,n,c) layout and accumulates
// per-channel sum/sumsq for BatchNorm via LDS + global atomics.
// One workgroup (8 waves) per 128-column tile.
// ---------------------------------------------------------------------------
#define NT   128
#define XSTR 144   // 128 + 16 pad  (keeps 32B fragment alignment, rotates banks)
#define HSTR 528   // 512 + 16 pad

#define WMMA_BF16(ACC, A, BPTR)                                               \
    do {                                                                      \
        v16bf bb_ = *(const v16bf*)(BPTR);                                    \
        ACC = __builtin_amdgcn_wmma_f32_16x16x32_bf16(                        \
            false, (A), false, bb_, (short)0, (ACC), false, false);           \
    } while (0)

__global__ __launch_bounds__(256) void ffn_kernel(const float* __restrict__ in,
                                                  const unsigned short* __restrict__ vws,
                                                  const unsigned short* __restrict__ w1s,
                                                  const unsigned short* __restrict__ w2s,
                                                  const float* __restrict__ b1,
                                                  const float* __restrict__ b2,
                                                  float* __restrict__ out2,
                                                  float* __restrict__ gsum,
                                                  float* __restrict__ gsumsq) {
    __shared__ unsigned short sX[NT][XSTR];   // [n][j]  bf16 ffn input, j-major
    __shared__ unsigned short sH[NT][HSTR];   // [n][k]  bf16 hidden, transposed
    __shared__ float sB1[DFFN_HID];
    __shared__ float sB2[DOUT];
    __shared__ float sSum[64], sSq[64];

    const int tid  = threadIdx.x;
    const int lane = tid & 31;
    const int wave = tid >> 5;
    const int nlo  = lane & 15;
    const int klo  = (lane >> 4) << 4;
    const int nt = blockIdx.x & 15;           // 2048/128 col tiles per (b,t)
    const int bt = blockIdx.x >> 4;
    const int b = bt / TT, t = bt % TT;
    const int n0 = nt * NT;
    const long cbase = (long)bt * NN;

    if (tid < 64) { sSum[tid] = 0.0f; sSq[tid] = 0.0f; }
    for (int i = tid; i < DFFN_HID; i += 256) sB1[i] = b1[i];
    if (tid < DOUT) sB2[tid] = b2[tid];

    // x channels (coalesced over j)
    for (int i = tid; i < NT * 64; i += 256) {
        int j = i & 63, n = i >> 6;
        sX[n][j] = f2bf(in[(cbase + n0 + n) * FF + j]);
    }
    __syncthreads();
    // v channels (coalesced over n): sX[n][j] = x - v ; sX[n][64+j] = v
    for (int i = tid; i < NT * 64; i += 256) {
        int n = i & 127, c = i >> 7;
        float vv = bf2f(vws[(((long)(b * 64 + c)) * TT + t) * NN + n0 + n]);
        float xv = bf2f(sX[n][c]);
        sX[n][c]      = f2bf(xv - vv);
        sX[n][64 + c] = f2bf(vv);
    }
    // warm this wave's W2 fragment block for GEMM2 while GEMM1 runs
    __builtin_prefetch((const char*)(w2s + (wave & 3) * 8192), 0, 2);
    __syncthreads();

    const int ctb = (wave >> 2) << 2;         // 4 column tiles per wave

    // ------------- GEMM1: H = gelu(W1 (512x128) * Xin (128x128) + b1) -------
    {
        const int mtb = (wave & 3) << 3;      // 8 row tiles per wave
        for (int rr = 0; rr < 8; ++rr) {
            const int mt = mtb + rr;
            v8f a0 = {0.f,0.f,0.f,0.f,0.f,0.f,0.f,0.f};
            v8f a1 = a0, a2 = a0, a3 = a0;
            #pragma unroll
            for (int kt = 0; kt < 4; ++kt) {  // A loaded once, 4 WMMAs issued
                v16bf a = *(const v16bf*)(w1s + ((((mt << 2) + kt) << 5) + lane) * 16);
                const int kof = (kt << 5) + klo;
                WMMA_BF16(a0, a, &sX[(ctb + 0) * 16 + nlo][kof]);
                WMMA_BF16(a1, a, &sX[(ctb + 1) * 16 + nlo][kof]);
                WMMA_BF16(a2, a, &sX[(ctb + 2) * 16 + nlo][kof]);
                WMMA_BF16(a3, a, &sX[(ctb + 3) * 16 + nlo][kof]);
            }
            #pragma unroll
            for (int r = 0; r < 8; ++r) {
                const int m = (mt << 4) + r + ((lane >> 4) << 3);
                const float bm = sB1[m];
                sH[(ctb + 0) * 16 + nlo][m] = f2bf(gelu_f(a0[r] + bm));
                sH[(ctb + 1) * 16 + nlo][m] = f2bf(gelu_f(a1[r] + bm));
                sH[(ctb + 2) * 16 + nlo][m] = f2bf(gelu_f(a2[r] + bm));
                sH[(ctb + 3) * 16 + nlo][m] = f2bf(gelu_f(a3[r] + bm));
            }
        }
    }
    __syncthreads();

    // ------------- GEMM2: Out = W2 (64x512) * H (512x128) + b2 + x ----------
    {
        const int mt = wave & 3;              // 4 row tiles of 16 channels
        v8f a0 = {0.f,0.f,0.f,0.f,0.f,0.f,0.f,0.f};
        v8f a1 = a0, a2 = a0, a3 = a0;
        #pragma unroll
        for (int kt = 0; kt < 16; ++kt) {
            v16bf a = *(const v16bf*)(w2s + ((((mt << 4) + kt) << 5) + lane) * 16);
            const int kof = (kt << 5) + klo;
            WMMA_BF16(a0, a, &sH[(ctb + 0) * 16 + nlo][kof]);
            WMMA_BF16(a1, a, &sH[(ctb + 1) * 16 + nlo][kof]);
            WMMA_BF16(a2, a, &sH[(ctb + 2) * 16 + nlo][kof]);
            WMMA_BF16(a3, a, &sH[(ctb + 3) * 16 + nlo][kof]);
        }
        #pragma unroll
        for (int i = 0; i < 4; ++i) {
            const v8f acc = (i == 0) ? a0 : (i == 1) ? a1 : (i == 2) ? a2 : a3;
            const int n = n0 + (ctb + i) * 16 + nlo;
            const long obase = (cbase + n) * (long)DOUT;
            #pragma unroll
            for (int r = 0; r < 8; ++r) {
                const int m = (mt << 4) + r + ((lane >> 4) << 3);
                float val = acc[r] + sB2[m] + in[obase + m];
                out2[obase + m] = val;
                atomicAdd(&sSum[m], val);
                atomicAdd(&sSq[m], val * val);
            }
        }
    }
    __syncthreads();
    if (tid < 64) {
        atomicAdd(&gsum[tid],   sSum[tid]);
        atomicAdd(&gsumsq[tid], sSq[tid]);
    }
}

// ---------------------------------------------------------------------------
// Kernel 3: fold BN stats into per-channel scale/shift.
// ---------------------------------------------------------------------------
__global__ void stats_kernel(const float* __restrict__ gsum,
                             const float* __restrict__ gsumsq,
                             const float* __restrict__ gamma,
                             const float* __restrict__ beta,
                             float* __restrict__ scsh) {
    int c = threadIdx.x;
    if (c < 64) {
        const float cnt = (float)BB * TT * NN;
        float mean = gsum[c] / cnt;
        float var  = gsumsq[c] / cnt - mean * mean;
        float inv  = rsqrtf(var + 1e-5f);
        float sc   = gamma[c] * inv;
        scsh[c]      = sc;
        scsh[64 + c] = beta[c] - mean * sc;
    }
}

// ---------------------------------------------------------------------------
// Kernel 4: normalize, fully coalesced float4; out2 layout == output layout.
// ---------------------------------------------------------------------------
__global__ __launch_bounds__(256) void finalize_kernel(const float* __restrict__ out2,
                                                       const float* __restrict__ scsh,
                                                       float* __restrict__ out) {
    __shared__ float s[128];
    if (threadIdx.x < 128) s[threadIdx.x] = scsh[threadIdx.x];
    __syncthreads();
    long i4 = ((long)blockIdx.x * 256 + threadIdx.x) * 4;
    const float4 v = *(const float4*)(out2 + i4);
    int c = (int)(i4 & 63);
    float4 r;
    r.x = v.x * s[c + 0] + s[64 + c + 0];
    r.y = v.y * s[c + 1] + s[64 + c + 1];
    r.z = v.z * s[c + 2] + s[64 + c + 2];
    r.w = v.w * s[c + 3] + s[64 + c + 3];
    *(float4*)(out + i4) = r;
}

// ---------------------------------------------------------------------------
// Launch
// ---------------------------------------------------------------------------
extern "C" void kernel_launch(void* const* d_in, const int* in_sizes, int n_in,
                              void* d_out, int out_size, void* d_ws, size_t ws_size,
                              hipStream_t stream) {
    const float* in    = (const float*)d_in[0];
    // d_in[1] = ssie (unused by forward)
    const float* cores = (const float*)d_in[2];
    const float* vw    = (const float*)d_in[3];
    const float* vb    = (const float*)d_in[4];
    const float* w1    = (const float*)d_in[5];
    const float* b1    = (const float*)d_in[6];
    const float* w2    = (const float*)d_in[7];
    const float* b2    = (const float*)d_in[8];
    const float* gamma = (const float*)d_in[9];
    const float* beta  = (const float*)d_in[10];

    // workspace layout (bytes, all 32B aligned):
    //   vws    : 16*64*12*2048 bf16      =  50,331,648
    //   w1s    : 512*128 bf16            =     131,072
    //   w2s    : 64*512 bf16             =      65,536
    //   gsum   : 64 f32  / gsumsq 64 f32 =         512
    //   scsh   : 128 f32                 =         512
    //   out2   : 16*12*2048*64 f32       = 100,663,296
    char* ws = (char*)d_ws;
    unsigned short* vws = (unsigned short*)ws;
    unsigned short* w1s = (unsigned short*)(ws + 50331648);
    unsigned short* w2s = (unsigned short*)(ws + 50462720);
    float* gsum   = (float*)(ws + 50528256);
    float* gsumsq = gsum + 64;
    float* scsh   = (float*)(ws + 50528768);
    float* out2   = (float*)(ws + 50529280);

    prep_kernel<<<384, 256, 0, stream>>>(w1, w2, w1s, w2s, gsum);
    attn_kernel<<<BB * HEAD * TT, 256, 0, stream>>>(in, cores, vw, vb, vws);
    ffn_kernel<<<BB * TT * (NN / NT), 256, 0, stream>>>(in, vws, w1s, w2s, b1, b2,
                                                        out2, gsum, gsumsq);
    stats_kernel<<<1, 64, 0, stream>>>(gsum, gsumsq, gamma, beta, scsh);
    finalize_kernel<<<(BB * TT * NN * DOUT) / 1024, 256, 0, stream>>>(out2, scsh,
                                                                      (float*)d_out);
}
